// MultiHeadAttention_27728308863314
// MI455X (gfx1250) — matrix operand order repaired and checked by reading, not compiled, not run on hardware
//
#include <hip/hip_runtime.h>
#include <hip/hip_bf16.h>

typedef __attribute__((ext_vector_type(16))) _Float16 v16h;
typedef __attribute__((ext_vector_type(8)))  _Float16 h8;
typedef __attribute__((ext_vector_type(8)))  float    v8f;

union FragAB { v16h v; h8 h2[2]; _Float16 h[16]; };
union FragC  { v8f v; float f[8]; };

#define WMMA_F16(a, b, c) \
  __builtin_amdgcn_wmma_f32_16x16x32_f16(false, (a), false, (b), (short)0, (c), false, false)

constexpr int S_     = 2048;
constexpr int HID    = 2048;
constexpr int NH     = 16;
constexpr int NKV    = 2;
constexpr int HD     = 128;
constexpr int VALID_ = 1792;

// ---------------------------------------------------------------------------
// f32 -> f16 conversion (row-major copy, used for x)
// ---------------------------------------------------------------------------
__global__ void cvt_f32_f16(const float* __restrict__ in, _Float16* __restrict__ out, int n) {
  int i = blockIdx.x * blockDim.x + threadIdx.x;
  if (i < n) out[i] = (_Float16)in[i];
}

// ---------------------------------------------------------------------------
// Tiled transpose + f32->f16: in[K][N] -> out[N][K].
// 32x32 LDS tile, blockDim (32,8). One-time pass; makes GEMM B-fragments
// contiguous 16B global loads (L2 is 192MB: whole weight set stays resident).
// ---------------------------------------------------------------------------
__global__ __launch_bounds__(256) void transpose_cvt(
    const float* __restrict__ in, _Float16* __restrict__ out, int K, int N)
{
  __shared__ float tile[32][33];
  const int n0 = blockIdx.x * 32;
  const int k0 = blockIdx.y * 32;
  #pragma unroll
  for (int i = threadIdx.y; i < 32; i += 8)
    tile[i][threadIdx.x] = in[(size_t)(k0 + i) * N + n0 + threadIdx.x];
  __syncthreads();
  #pragma unroll
  for (int i = threadIdx.y; i < 32; i += 8)
    out[(size_t)(n0 + i) * K + k0 + threadIdx.x] = (_Float16)tile[threadIdx.x][i];
}

// ---------------------------------------------------------------------------
// WMMA GEMM, B pre-transposed: C[M,N] (f32) = A[M,K] (f16) @ Bt[N,K]^T + bias
// block = 128 threads (4 waves); wave tile 16x64; no LDS, no barriers.
// ---------------------------------------------------------------------------
__global__ __launch_bounds__(128) void gemm_wmma_nt(
    const _Float16* __restrict__ A, const _Float16* __restrict__ Bt,
    const float* __restrict__ bias, float* __restrict__ C,
    int M, int N, int K)
{
  const int lane    = threadIdx.x & 31;
  const int wave    = threadIdx.x >> 5;
  const int half_id = lane >> 4;
  const int l16     = lane & 15;
  const int row0    = blockIdx.y * 64 + wave * 16;
  const int col0    = blockIdx.x * 64;

  FragC acc[4];
  #pragma unroll
  for (int i = 0; i < 4; i++)
    #pragma unroll
    for (int e = 0; e < 8; e++) acc[i].f[e] = 0.0f;

  const _Float16* ap  = A + (size_t)(row0 + l16) * K;
  const _Float16* bp0 = Bt + (size_t)(col0 + 0 * 16 + l16) * K;
  const _Float16* bp1 = Bt + (size_t)(col0 + 1 * 16 + l16) * K;
  const _Float16* bp2 = Bt + (size_t)(col0 + 2 * 16 + l16) * K;
  const _Float16* bp3 = Bt + (size_t)(col0 + 3 * 16 + l16) * K;
  const _Float16* bps[4] = { bp0, bp1, bp2, bp3 };

  for (int k0 = 0; k0 < K; k0 += 32) {
    __builtin_prefetch(ap + k0 + 32, 0, 1);
    __builtin_prefetch(bp0 + k0 + 32, 0, 1);

    FragAB af;   // ISA 16-bit A layout: two 8-half runs per lane
    af.h2[0] = *(const h8*)(ap + k0 + half_id * 8);
    af.h2[1] = *(const h8*)(ap + k0 + 16 + half_id * 8);

    #pragma unroll
    for (int nt = 0; nt < 4; nt++) {
      FragAB bf;  // B layout: lane's column = l16; two 8-half runs per lane (same K order as A)
      bf.h2[0] = *(const h8*)(bps[nt] + k0 + half_id * 8);
      bf.h2[1] = *(const h8*)(bps[nt] + k0 + 16 + half_id * 8);
      acc[nt].v = WMMA_F16(af.v, bf.v, acc[nt].v);
    }
  }

  #pragma unroll
  for (int nt = 0; nt < 4; nt++)
    #pragma unroll
    for (int r = 0; r < 8; r++) {
      int row = row0 + r + half_id * 8;   // C layout: VGPR r -> M=r / r+8
      int col = col0 + nt * 16 + l16;
      float v = acc[nt].f[r];
      if (bias) v += bias[col];
      C[(size_t)row * N + col] = v;
    }
}

// ---------------------------------------------------------------------------
// RoPE (sectioned reorder) + mask + f16 conversion; also writes V transposed.
// ---------------------------------------------------------------------------
__global__ __launch_bounds__(128) void rope_mask_cvt(
    const float* __restrict__ Qf, const float* __restrict__ Kf, const float* __restrict__ Vf,
    const float* __restrict__ cosb, const float* __restrict__ sinb, const float* __restrict__ mask,
    _Float16* __restrict__ Qh, _Float16* __restrict__ Kh, _Float16* __restrict__ Vt)
{
  const int s = blockIdx.x;
  const int d = threadIdx.x;
  const int type = (d < 16) ? 0 : (d < 40) ? 1 : (d < 64) ? 2
                 : (d < 80) ? 0 : (d < 104) ? 1 : 2;
  const float ce  = cosb[((size_t)type * S_ + s) * HD + d];
  const float se  = sinb[((size_t)type * S_ + s) * HD + d];
  const float mk  = mask[s];
  const int   dro = (d < 64) ? d + 64 : d - 64;
  const float sgn = (d < 64) ? -1.0f : 1.0f;

  #pragma unroll
  for (int hh = 0; hh < NH; hh++) {
    float q  = Qf[(size_t)s * (NH * HD) + hh * HD + d];
    float qo = Qf[(size_t)s * (NH * HD) + hh * HD + dro];
    Qh[(size_t)s * (NH * HD) + hh * HD + d] = (_Float16)(q * ce + sgn * qo * se);
  }
  #pragma unroll
  for (int c = 0; c < NKV; c++) {
    float kx = Kf[(size_t)s * (NKV * HD) + c * HD + d];
    float ko = Kf[(size_t)s * (NKV * HD) + c * HD + dro];
    Kh[(size_t)s * (NKV * HD) + c * HD + d] = (_Float16)((kx * ce + sgn * ko * se) * mk);
    float vv = Vf[(size_t)s * (NKV * HD) + c * HD + d] * mk;
    Vt[(size_t)(c * HD + d) * S_ + s] = (_Float16)vv;
  }
}

// ---------------------------------------------------------------------------
// Flash attention: one wave per (head, 16-query tile). Online softmax.
// K tiles double-buffered in LDS via GLOBAL_LOAD_ASYNC_TO_LDS_B128 (ASYNCcnt):
// async engine copies chunk j+32 while WMMAs crunch chunk j.
// ---------------------------------------------------------------------------
__global__ __launch_bounds__(32) void attn_fa(
    const _Float16* __restrict__ Qh, const _Float16* __restrict__ Kh,
    const _Float16* __restrict__ Vt, _Float16* __restrict__ Oh)
{
  __shared__ __align__(16) _Float16 Ps[16][40];
  __shared__ __align__(16) _Float16 Kb[2][32][136];   // 272B row stride: de-conflicts banks
  const int lane    = threadIdx.x & 31;
  const int half_id = lane >> 4;
  const int l16     = lane & 15;
  const int qb      = blockIdx.x * 16;
  const int h       = blockIdx.y;
  const int kv      = h >> 3;              // rep = H/KV = 8

  // Q A-fragments: 16 queries x 128 d, 4 chunks of K=32
  FragAB qf[4];
  {
    const _Float16* qp = Qh + (size_t)(qb + l16) * (NH * HD) + h * HD;
    #pragma unroll
    for (int dc = 0; dc < 4; dc++) {
      qf[dc].h2[0] = *(const h8*)(qp + dc * 32 + half_id * 8);
      qf[dc].h2[1] = *(const h8*)(qp + dc * 32 + 16 + half_id * 8);
    }
  }

  // async stage of 32 K rows (8KB) into LDS buffer: 16 ops x 32 lanes x 16B
  auto stage_k = [&](int j, int buf) {
    #pragma unroll
    for (int it = 0; it < 16; it++) {
      int idx = it * 32 + lane;
      int row = idx >> 4;
      int seg = idx & 15;
      const _Float16* g = Kh + (size_t)(j + row) * (NKV * HD) + kv * HD + seg * 8;
      unsigned lds = (unsigned)(uintptr_t)&Kb[buf][row][seg * 8];  // LDS addr = flat[31:0]
      asm volatile("global_load_async_to_lds_b128 %0, %1, off"
                   :: "v"(lds), "v"((unsigned long long)(uintptr_t)g) : "memory");
    }
  };

  float mi[8], li[8];
  FragC acc[8];
  #pragma unroll
  for (int r = 0; r < 8; r++) { mi[r] = -3.0e38f; li[r] = 0.0f; }
  #pragma unroll
  for (int nt = 0; nt < 8; nt++)
    #pragma unroll
    for (int e = 0; e < 8; e++) acc[nt].f[e] = 0.0f;

  const float scl = 0.08838834764831845f;  // D^-0.5
  const int jend = (qb + 16 < VALID_) ? qb + 16 : VALID_;  // causal + valid-length bound

  stage_k(0, 0);                            // prologue: chunk 0 -> buffer 0
  int buf = 0;

  for (int j = 0; j < jend; j += 32) {
    const bool have_next = (j + 32) < jend;
    if (have_next) {
      stage_k(j + 32, buf ^ 1);             // overlap copy with compute
      asm volatile("s_wait_asynccnt 0x10" ::: "memory");   // retire older 16 only
    } else {
      asm volatile("s_wait_asynccnt 0x0" ::: "memory");
    }
    __builtin_prefetch(Vt + (size_t)(kv * HD + l16) * S_ + j, 0, 1);

    // scores: two 16x16 subtiles, contraction over D via 4 WMMAs each
    FragC sc[2];
    #pragma unroll
    for (int sub = 0; sub < 2; sub++) {
      #pragma unroll
      for (int e = 0; e < 8; e++) sc[sub].f[e] = 0.0f;
      const _Float16* kp = &Kb[buf][sub * 16 + l16][0];   // LDS-resident K rows
      #pragma unroll
      for (int dc = 0; dc < 4; dc++) {
        FragAB bf;  // B = K^T: lane's column = key (l16), elements = two 8-d runs (same order as A)
        bf.h2[0] = *(const h8*)(kp + dc * 32 + half_id * 8);
        bf.h2[1] = *(const h8*)(kp + dc * 32 + 16 + half_id * 8);
        sc[sub].v = WMMA_F16(qf[dc].v, bf.v, sc[sub].v);
      }
    }

    // mask + scale + online softmax (per-lane row stats; rows map to VGPR r)
    #pragma unroll
    for (int r = 0; r < 8; r++) {
      const int qrow = qb + r + half_id * 8;
      #pragma unroll
      for (int sub = 0; sub < 2; sub++) {
        int key = j + sub * 16 + l16;
        float sv = sc[sub].f[r] * scl;
        sc[sub].f[r] = (key > qrow || key >= VALID_) ? -1.0e9f : sv;
      }
      float rm = fmaxf(sc[0].f[r], sc[1].f[r]);
      rm = fmaxf(rm, __shfl_xor(rm, 1, 32));
      rm = fmaxf(rm, __shfl_xor(rm, 2, 32));
      rm = fmaxf(rm, __shfl_xor(rm, 4, 32));
      rm = fmaxf(rm, __shfl_xor(rm, 8, 32));
      float newm = fmaxf(mi[r], rm);
      float p0 = __expf(sc[0].f[r] - newm);
      float p1 = __expf(sc[1].f[r] - newm);
      sc[0].f[r] = p0; sc[1].f[r] = p1;
      float ps = p0 + p1;
      ps += __shfl_xor(ps, 1, 32);
      ps += __shfl_xor(ps, 2, 32);
      ps += __shfl_xor(ps, 4, 32);
      ps += __shfl_xor(ps, 8, 32);
      float corr = __expf(mi[r] - newm);
      li[r] = li[r] * corr + ps;
      mi[r] = newm;
      #pragma unroll
      for (int nt = 0; nt < 8; nt++) acc[nt].f[r] *= corr;
    }

    // C-layout -> A-layout for P via LDS round trip
    #pragma unroll
    for (int sub = 0; sub < 2; sub++)
      #pragma unroll
      for (int r = 0; r < 8; r++)
        Ps[r + half_id * 8][sub * 16 + l16] = (_Float16)sc[sub].f[r];
    __syncthreads();
    FragAB pf;
    pf.h2[0] = *(const h8*)&Ps[l16][half_id * 8];
    pf.h2[1] = *(const h8*)&Ps[l16][16 + half_id * 8];
    __syncthreads();

    // P @ V: 8 output n-tiles; V^T gives contiguous B-fragment reads
    #pragma unroll
    for (int nt = 0; nt < 8; nt++) {
      const _Float16* vp = Vt + (size_t)(kv * HD + nt * 16 + l16) * S_ + j + half_id * 8;
      FragAB bf;
      bf.h2[0] = *(const h8*)vp;
      bf.h2[1] = *(const h8*)(vp + 16);
      acc[nt].v = WMMA_F16(pf.v, bf.v, acc[nt].v);
    }
    buf ^= 1;
  }

  // normalize and store f16 for the output projection GEMM
  #pragma unroll
  for (int nt = 0; nt < 8; nt++)
    #pragma unroll
    for (int r = 0; r < 8; r++) {
      int row = qb + r + half_id * 8;
      float inv = 1.0f / li[r];
      Oh[(size_t)row * (NH * HD) + h * HD + nt * 16 + l16] = (_Float16)(acc[nt].f[r] * inv);
    }
}

// ---------------------------------------------------------------------------
// Launch
// ---------------------------------------------------------------------------
extern "C" void kernel_launch(void* const* d_in, const int* in_sizes, int n_in,
                              void* d_out, int out_size, void* d_ws, size_t ws_size,
                              hipStream_t stream) {
  const float* x    = (const float*)d_in[0];
  const float* cosb = (const float*)d_in[1];
  const float* sinb = (const float*)d_in[2];
  const float* mask = (const float*)d_in[3];
  const float* Wq   = (const float*)d_in[4];
  const float* bq   = (const float*)d_in[5];
  const float* Wk   = (const float*)d_in[6];
  const float* bk   = (const float*)d_in[7];
  const float* Wv   = (const float*)d_in[8];
  const float* bv   = (const float*)d_in[9];
  const float* Wo   = (const float*)d_in[10];
  float* out = (float*)d_out;

  char* w = (char*)d_ws;                      // 64 MB layout, all offsets 16B-aligned
  _Float16* xh   = (_Float16*)(w + 0);         //  8 MB  x          [S][HID] f16
  _Float16* Wqt  = (_Float16*)(w + 8388608);   //  8 MB  Wq^T       [2048][2048] f16
  _Float16* Wkt  = (_Float16*)(w + 16777216);  //  1 MB  Wk^T       [256][2048] f16
  _Float16* Wvt  = (_Float16*)(w + 17825792);  //  1 MB  Wv^T       [256][2048] f16
  _Float16* Wot  = (_Float16*)(w + 18874368);  //  8 MB  Wo^T       [2048][2048] f16
  float*    Qf   = (float*)   (w + 27262976);  // 16 MB  Q f32      [S][2048]
  float*    Kf   = (float*)   (w + 44040192);  //  2 MB  K f32      [S][256]
  float*    Vf   = (float*)   (w + 46137344);  //  2 MB  V f32      [S][256]
  _Float16* Qh   = (_Float16*)(w + 48234496);  //  8 MB  Q roped    [S][2048] f16
  _Float16* Kh   = (_Float16*)(w + 56623104);  //  1 MB  K roped    [S][256] f16
  _Float16* Vt   = (_Float16*)(w + 57671680);  //  1 MB  V^T        [256][S] f16
  _Float16* Ah   = (_Float16*)(w + 58720256);  //  8 MB  attn out   [S][2048] f16

  cvt_f32_f16<<<(S_ * HID + 255) / 256, 256, 0, stream>>>(x, xh, S_ * HID);

  transpose_cvt<<<dim3((NH * HD) / 32, HID / 32), dim3(32, 8), 0, stream>>>(Wq, Wqt, HID, NH * HD);
  transpose_cvt<<<dim3((NKV * HD) / 32, HID / 32), dim3(32, 8), 0, stream>>>(Wk, Wkt, HID, NKV * HD);
  transpose_cvt<<<dim3((NKV * HD) / 32, HID / 32), dim3(32, 8), 0, stream>>>(Wv, Wvt, HID, NKV * HD);
  transpose_cvt<<<dim3(HID / 32, (NH * HD) / 32), dim3(32, 8), 0, stream>>>(Wo, Wot, NH * HD, HID);

  gemm_wmma_nt<<<dim3((NH * HD) / 64, S_ / 64), 128, 0, stream>>>(xh, Wqt, bq, Qf, S_, NH * HD, HID);
  gemm_wmma_nt<<<dim3((NKV * HD) / 64, S_ / 64), 128, 0, stream>>>(xh, Wkt, bk, Kf, S_, NKV * HD, HID);
  gemm_wmma_nt<<<dim3((NKV * HD) / 64, S_ / 64), 128, 0, stream>>>(xh, Wvt, bv, Vf, S_, NKV * HD, HID);

  rope_mask_cvt<<<S_, HD, 0, stream>>>(Qf, Kf, Vf, cosb, sinb, mask, Qh, Kh, Vt);

  attn_fa<<<dim3(S_ / 16, NH), 32, 0, stream>>>(Qh, Kh, Vt, Ah);

  gemm_wmma_nt<<<dim3(HID / 64, S_ / 64), 128, 0, stream>>>(Ah, Wot, nullptr, out, S_, HID, NH * HD);
}
